// connectLoss_80350248174167
// MI455X (gfx1250) — compile-verified, matrix operand order checked by reading, not run on hardware
//
#include <hip/hip_runtime.h>
#include <hip/hip_bf16.h>

#define HH 768
#define WW 768
#define HW 589824            // 768*768
#define NB 8
#define NIMG 16              // images 0..7 = anno, 8..15 = pred argmax
#define IDXMASK 0x00FFFFFF
#define CLSMASK 0x7F000000
#define SENT    0x7FFFFFFF

typedef __attribute__((ext_vector_type(8))) int v8i;

__device__ __forceinline__ int encodeLab(int cls, int i) {
    return (cls == 2 || cls == 3) ? ((cls << 24) | i) : SENT;
}

// ---------------------------------------------------------------------------
// Kernel 1: build label maps, 4 pixels per thread via B128 loads/stores.
// img < 8 : class = anno[img][i]
// img >= 8: class = argmax_c pred[img-8][c][i]  (first max wins, like jnp.argmax)
// ---------------------------------------------------------------------------
__global__ void k_init(const float* __restrict__ pred, const int* __restrict__ anno,
                       int* __restrict__ labels) {
    const int i0  = (blockIdx.x * blockDim.x + threadIdx.x) * 4;  // 0..HW-4 exact
    const int img = blockIdx.y;
    int4 out;
    if (img < NB) {
        const int4 a = *(const int4*)(anno + (size_t)img * HW + i0);
        out.x = encodeLab(a.x, i0 + 0);
        out.y = encodeLab(a.y, i0 + 1);
        out.z = encodeLab(a.z, i0 + 2);
        out.w = encodeLab(a.w, i0 + 3);
    } else {
        const int b = img - NB;
        const float* base = pred + (size_t)(b * 4) * HW + i0;
        __builtin_prefetch(base + 8192, 0, 0);                    // global_prefetch_b8
        const float4 v0 = *(const float4*)(base);
        const float4 v1 = *(const float4*)(base + (size_t)HW);
        const float4 v2 = *(const float4*)(base + (size_t)2 * HW);
        const float4 v3 = *(const float4*)(base + (size_t)3 * HW);
        float bst[4] = {v0.x, v0.y, v0.z, v0.w};
        int   cls[4] = {0, 0, 0, 0};
        const float c1[4] = {v1.x, v1.y, v1.z, v1.w};
        const float c2[4] = {v2.x, v2.y, v2.z, v2.w};
        const float c3[4] = {v3.x, v3.y, v3.z, v3.w};
#pragma unroll
        for (int k = 0; k < 4; ++k) {
            if (c1[k] > bst[k]) { bst[k] = c1[k]; cls[k] = 1; }   // strict > keeps
            if (c2[k] > bst[k]) { bst[k] = c2[k]; cls[k] = 2; }   // first max, like
            if (c3[k] > bst[k]) { bst[k] = c3[k]; cls[k] = 3; }   // jnp.argmax
        }
        out.x = encodeLab(cls[0], i0 + 0);
        out.y = encodeLab(cls[1], i0 + 1);
        out.z = encodeLab(cls[2], i0 + 2);
        out.w = encodeLab(cls[3], i0 + 3);
    }
    *(int4*)(labels + (size_t)img * HW + i0) = out;
}

__global__ void k_zero(int* __restrict__ counts) {
    if (threadIdx.x < 2 * NIMG) counts[threadIdx.x] = 0;
}

// ---------------------------------------------------------------------------
// Kernel 2: lock-free union-find (atomicMin linking + path halving).
// Parent pointers strictly decrease; class/sentinel bits are invariant along
// any chain (unions only between same-class pixels), so full-word atomicMin
// == index-min, and stale eligibility reads are safe.
// ---------------------------------------------------------------------------
__device__ __forceinline__ int ldL(const int* L, int i) {
    return __hip_atomic_load(L + i, __ATOMIC_RELAXED, __HIP_MEMORY_SCOPE_AGENT);
}

__device__ int findRoot(int* L, int i, int clsb) {
    int p = ldL(L, i) & IDXMASK;
    while (p != i) {
        int gp = ldL(L, p) & IDXMASK;
        if (gp != p) atomicMin(&L[i], clsb | gp);            // path halving (monotone)
        i = p;
        p = gp;
    }
    return i;
}

__device__ void unite(int* L, int a, int b, int clsb) {
    int ra = findRoot(L, a, clsb);
    int rb = findRoot(L, b, clsb);
    while (ra != rb) {
        const int mn = ra < rb ? ra : rb;
        const int mx = ra ^ rb ^ mn;
        const int old = atomicMin(&L[mx], clsb | mn);
        const int oi  = old & IDXMASK;
        if (oi == mx) return;                                // mx was root: linked
        ra = findRoot(L, oi, clsb);
        rb = findRoot(L, mn, clsb);
    }
}

__device__ __forceinline__ bool compat(int a, int b) {
    return a != SENT && b != SENT && ((a & CLSMASK) == (b & CLSMASK));
}

// One thread per 4-pixel row strip (768 % 4 == 0: a quad never crosses a row).
__global__ void k_merge(int* __restrict__ labels) {
    const int i0 = (blockIdx.x * blockDim.x + threadIdx.x) * 4;
    int* L = labels + (size_t)blockIdx.y * HW;
    __builtin_prefetch(&L[(i0 + 8192 < HW) ? (i0 + 8192) : i0], 0, 0);
    const int4 mv = *(const int4*)(L + i0);
    const int m[4] = {mv.x, mv.y, mv.z, mv.w};
    const int x0 = i0 % WW;
    // horizontal inside the quad
#pragma unroll
    for (int k = 0; k < 3; ++k)
        if (compat(m[k], m[k + 1])) unite(L, i0 + k, i0 + k + 1, m[k] & CLSMASK);
    // horizontal to next quad
    if (x0 + 4 < WW) {
        const int r = L[i0 + 4];
        if (compat(m[3], r)) unite(L, i0 + 3, i0 + 4, m[3] & CLSMASK);
    }
    // vertical to the row below
    if (i0 + WW < HW) {
        const int4 bv = *(const int4*)(L + i0 + WW);
        const int bl[4] = {bv.x, bv.y, bv.z, bv.w};
#pragma unroll
        for (int k = 0; k < 4; ++k)
            if (compat(m[k], bl[k])) unite(L, i0 + k, i0 + WW + k, m[k] & CLSMASK);
    }
}

// ---------------------------------------------------------------------------
// Kernel 3: count roots per (image, class) with V_WMMA_I32_16X16X64_IU8.
// Indicator bytes -> A (16x64 u8 striped over the wave), B = ones:
// D[m][n] = C[m][n] + sum_k A[m][k]  =>  sum over all of D = 16 * (#indicators).
// B128 label loads: one int4 per lane per j -> one packed u32 word.
// Grid: (9, 16) x 256 threads -> 72 waves/image, 8192 px/wave, 8 WMMA steps.
// ---------------------------------------------------------------------------
__global__ void k_count(const int* __restrict__ labels, int* __restrict__ counts) {
    const int img  = blockIdx.y;
    const int* L   = labels + (size_t)img * HW;
    const int lane = threadIdx.x & 31;
    const int wave = threadIdx.x >> 5;
    const int wid  = blockIdx.x * 8 + wave;                  // 0..71
    const int base = wid * 8192;

    v8i ones;
#pragma unroll
    for (int r = 0; r < 8; ++r) ones[r] = 0x01010101;
    v8i acc2 = {};
    v8i acc3 = {};

    for (int it = 0; it < 8; ++it) {
        const int pbase = base + it * 1024;
        unsigned w2[8], w3[8];
#pragma unroll
        for (int j = 0; j < 8; ++j) {
            const int pix0 = pbase + j * 128 + lane * 4;     // coalesced 512B/wave
            const int4 lv = *(const int4*)(L + pix0);
            const int lab[4] = {lv.x, lv.y, lv.z, lv.w};
            unsigned p2w = 0u, p3w = 0u;
#pragma unroll
            for (int k = 0; k < 4; ++k) {
                const int p = pix0 + k;
                const bool isRoot = (lab[k] != SENT) && ((lab[k] & IDXMASK) == p);
                const unsigned cls = ((unsigned)lab[k] >> 24) & 0x7Fu;
                p2w |= ((isRoot && cls == 2u) ? 1u : 0u) << (k * 8);
                p3w |= ((isRoot && cls == 3u) ? 1u : 0u) << (k * 8);
            }
            w2[j] = p2w;
            w3[j] = p3w;
        }
        v8i a2, a3;
#pragma unroll
        for (int r = 0; r < 8; ++r) { a2[r] = (int)w2[r]; a3[r] = (int)w3[r]; }
        // 7 args: (sgn_a, A, sgn_b, B, C, reuse_a, reuse_b)
        acc2 = __builtin_amdgcn_wmma_i32_16x16x64_iu8(false, a2, false, ones, acc2, false, false);
        acc3 = __builtin_amdgcn_wmma_i32_16x16x64_iu8(false, a3, false, ones, acc3, false, false);
    }

    int s2 = 0, s3 = 0;
#pragma unroll
    for (int r = 0; r < 8; ++r) { s2 += acc2[r]; s3 += acc3[r]; }
    for (int off = 16; off > 0; off >>= 1) {                 // wave32 butterfly
        s2 += __shfl_xor(s2, off, 32);
        s3 += __shfl_xor(s3, off, 32);
    }
    if (lane == 0) {
        atomicAdd(&counts[img * 2 + 0], s2 >> 4);            // /16 columns
        atomicAdd(&counts[img * 2 + 1], s3 >> 4);
    }
}

// ---------------------------------------------------------------------------
// Kernel 4: final scalar (matches reference f32 arithmetic).
// ---------------------------------------------------------------------------
__global__ void k_final(const int* __restrict__ counts, float* __restrict__ out) {
    if (threadIdx.x != 0) return;
    float acc = 0.0f;
    for (int b = 0; b < NB; ++b) {
        const float a2 = (float)counts[b * 2 + 0];
        const float a3 = (float)counts[b * 2 + 1];
        const float p2 = (float)counts[(NB + b) * 2 + 0];
        const float p3 = (float)counts[(NB + b) * 2 + 1];
        acc += (1.0f - 2.0f * fminf(p2, a2) / (a2 + p2));
        acc += (1.0f - 2.0f * fminf(p3, a3) / (a3 + p3));
    }
    *out = acc * 0.5f;
}

extern "C" void kernel_launch(void* const* d_in, const int* in_sizes, int n_in,
                              void* d_out, int out_size, void* d_ws, size_t ws_size,
                              hipStream_t stream) {
    (void)in_sizes; (void)n_in; (void)out_size; (void)ws_size;
    const float* pred = (const float*)d_in[0];   // (8,4,1,768,768) f32
    const int*   anno = (const int*)d_in[1];     // (8,1,768,768)   i32
    float* out = (float*)d_out;                  // scalar f32

    int* counts = (int*)d_ws;                          // 32 ints
    int* labels = (int*)((char*)d_ws + 256);           // 16 * 589824 ints (~37.7 MB)

    const dim3 blk(256);                               // 8 wave32s
    const dim3 grd4(HW / 1024, NIMG);                  // 576 x 16, 4 px/thread, exact

    k_init <<<grd4, blk, 0, stream>>>(pred, anno, labels);
    k_zero <<<1, 64, 0, stream>>>(counts);
    k_merge<<<grd4, blk, 0, stream>>>(labels);
    k_count<<<dim3(9, NIMG), blk, 0, stream>>>(labels, counts);
    k_final<<<1, 32, 0, stream>>>(counts, out);
}
